// PennyLanePQC_9311489098344
// MI455X (gfx1250) — matrix operand order repaired and checked by reading, not compiled, not run on hardware
//
#include <hip/hip_runtime.h>

typedef __attribute__((ext_vector_type(16))) _Float16 v16h;
typedef __attribute__((ext_vector_type(8)))  float    v8f;

#define NQ        13
#define D1        (1 << NQ)     // 8192
#define NQ2       14
#define D2        (1 << NQ2)    // 16384
#define REPSN     2
#define NEG_SLOPE 0.01f
#define EPSV      1e-8f
#define BLOCK     256
#define NWAVES    (BLOCK / 32)
#define WROW      234           // all_weights row stride (3*p_len, p_len=78)

struct SMEM {
    float u3r[NQ2][2][2];
    float u3i[NQ2][2][2];
    float gr[3][16][16];   // 16x16 group unitaries (real)
    float gi[3][16][16];   // (imag)
    float red;
    float ex[NQ];
};

// ---- per-qubit U3 gate matrices: u = [[c, -e^{i la} s],[e^{i ph} s, e^{i(ph+la)} c]]
__device__ inline void computeU3(const float* wl, int n, SMEM& sm) {
    int i = threadIdx.x;
    if (i < n) {
        float th = wl[3 * i], ph = wl[3 * i + 1], la = wl[3 * i + 2];
        float c = __cosf(0.5f * th), s = __sinf(0.5f * th);
        float cl = __cosf(la),  sl = __sinf(la);
        float cp = __cosf(ph),  sp = __sinf(ph);
        float cq = __cosf(ph + la), sq = __sinf(ph + la);
        sm.u3r[i][0][0] = c;        sm.u3i[i][0][0] = 0.f;
        sm.u3r[i][0][1] = -cl * s;  sm.u3i[i][0][1] = -sl * s;
        sm.u3r[i][1][0] = cp * s;   sm.u3i[i][1][0] = sp * s;
        sm.u3r[i][1][1] = cq * c;   sm.u3i[i][1][1] = sq * c;
    }
}

// ---- Kronecker-fuse 4 single-qubit gates into 16x16 unitaries.
// Group g covers element strides 16^g * {1,2,4,8}; qubit for bit k is q = n-1-4g-k.
__device__ inline void buildGroups(int n, SMEM& sm) {
    int ng = (n - 1) >> 2;           // 3 full groups for n=13 and n=14
    int e = threadIdx.x, r = e >> 4, c = e & 15;
    for (int g = 0; g < ng; ++g) {
        float pr = 1.f, pi = 0.f;
#pragma unroll
        for (int k = 0; k < 4; ++k) {
            int q = n - 1 - 4 * g - k;
            int br = (r >> k) & 1, bc = (c >> k) & 1;
            float ur = sm.u3r[q][br][bc], ui = sm.u3i[q][br][bc];
            float nr = pr * ur - pi * ui;
            pi = pr * ui + pi * ur;
            pr = nr;
        }
        sm.gr[g][r][c] = pr;
        sm.gi[g][r][c] = pi;
    }
}

// ---- Apply one 16x16 complex group unitary to the state via WMMA.
// Complex trick: Cr = [Ur | -Ui] x [Sr; Si],  Ci = [Ui | Ur] x [Sr; Si]  (K=32, exact fit).
__device__ inline void applyGroup(int g, int n, float* sRe, float* sIm, SMEM& sm) {
    const int lane = threadIdx.x & 31;
    const int wv   = threadIdx.x >> 5;
    const int M    = lane & 15;
    const bool hiL = lane >= 16;

    // A fragments (16x32 f16). Lanes 0-15: element h -> K=h (0..15).
    // Lanes 16-31: element h -> K=16+h (16..31).
    v16h acr, aci;
#pragma unroll
    for (int h = 0; h < 16; ++h) {
        int v = h >> 1, pos = h & 1;
        int K = (hiL ? 16 : 0) + 2 * v + pos;
        if (K < 16) {   // real block
            acr[h] = (_Float16)sm.gr[g][M][K];
            aci[h] = (_Float16)sm.gi[g][M][K];
        } else {        // imag block
            acr[h] = (_Float16)(-sm.gi[g][M][K - 16]);
            aci[h] = (_Float16)( sm.gr[g][M][K - 16]);
        }
    }

    const int sh      = 4 * g;
    const int lowmask = (1 << sh) - 1;
    const int tiles   = 1 << (n - 8);           // (2^n/16)/16 ; 32 (n=13) or 64 (n=14)
    const float* bsrc = hiL ? sIm : sRe;        // B: rows 0-15 = Sr, rows 16-31 = Si

    for (int t = wv; t < tiles; t += NWAVES) {  // uniform trip count -> EXEC all ones
        int j    = (t << 4) | M;                // column = virtual chunk index
        int base = ((j >> sh) << (sh + 4)) | (j & lowmask);
        v16h b;
#pragma unroll
        for (int h = 0; h < 16; ++h)            // B 32x16: lane l half h -> K=(l<16?h:16+h)
            b[h] = (_Float16)bsrc[base | (h << sh)];
        v8f cr = {}, ci = {};
        cr = __builtin_amdgcn_wmma_f32_16x16x32_f16(false, acr, false, b, (short)0, cr, false, false);
        ci = __builtin_amdgcn_wmma_f32_16x16x32_f16(false, aci, false, b, (short)0, ci, false, false);
        int mb = hiL ? 8 : 0;                   // D: lane l VGPR v -> M = v + (l<16?0:8)
#pragma unroll
        for (int v = 0; v < 8; ++v) {
            int a = base | ((v + mb) << sh);
            sRe[a] = cr[v];
            sIm[a] = ci[v];
        }
    }
}

// ---- Leftover high-order qubits (stride >= 16^3): classic 2x2 pair update.
__device__ inline void applyQubit(int q, int n, float* sRe, float* sIm, SMEM& sm) {
    int m = 1 << (n - 1 - q);
    float u00r = sm.u3r[q][0][0], u00i = sm.u3i[q][0][0];
    float u01r = sm.u3r[q][0][1], u01i = sm.u3i[q][0][1];
    float u10r = sm.u3r[q][1][0], u10i = sm.u3i[q][1][0];
    float u11r = sm.u3r[q][1][1], u11i = sm.u3i[q][1][1];
    int half = 1 << (n - 1);
    for (int p = threadIdx.x; p < half; p += BLOCK) {
        int i0 = ((p & ~(m - 1)) << 1) | (p & (m - 1));
        int i1 = i0 | m;
        float ar = sRe[i0], ai = sIm[i0], br = sRe[i1], bi = sIm[i1];
        sRe[i0] = u00r * ar - u00i * ai + u01r * br - u01i * bi;
        sIm[i0] = u00r * ai + u00i * ar + u01r * bi + u01i * br;
        sRe[i1] = u10r * ar - u10i * ai + u11r * br - u11i * bi;
        sIm[i1] = u10r * ai + u10i * ar + u11r * bi + u11i * br;
    }
}

// ---- CNOT: swap amplitudes with (control=1, target) <-> (control=1, target^1)
__device__ inline void applyCNOT(int c, int t, int n, float* sRe, float* sIm) {
    int cm = 1 << (n - 1 - c), tm = 1 << (n - 1 - t);
    int mlo = cm < tm ? cm : tm;
    int mhi = cm < tm ? tm : cm;
    int quarter = 1 << (n - 2);
    for (int p = threadIdx.x; p < quarter; p += BLOCK) {
        int i = ((p & ~(mlo - 1)) << 1) | (p & (mlo - 1));
        i = ((i & ~(mhi - 1)) << 1) | (i & (mhi - 1));
        i |= cm;            // control = 1, target = 0
        int j = i | tm;
        float tr = sRe[i]; sRe[i] = sRe[j]; sRe[j] = tr;
        float ti = sIm[i]; sIm[i] = sIm[j]; sIm[j] = ti;
    }
}

__device__ void layer(const float* wl, int n, float* sRe, float* sIm, SMEM& sm) {
    computeU3(wl, n, sm);
    __syncthreads();
    buildGroups(n, sm);
    __syncthreads();
    int ng = (n - 1) >> 2;
    for (int g = 0; g < ng; ++g) {
        applyGroup(g, n, sRe, sIm, sm);   // tiles are disjoint -> no intra-group barrier
        __syncthreads();
    }
    for (int q = n - 4 * ng - 1; q >= 0; --q) {  // remaining 1 (n=13) / 2 (n=14) qubits
        applyQubit(q, n, sRe, sIm, sm);
        __syncthreads();
    }
    for (int i = 0; i + 1 < n; i += 2) { applyCNOT(i, i + 1, n, sRe, sIm); __syncthreads(); }
    for (int i = 1; i < n; i += 2) {
        int t = (i + 1 == n) ? 0 : (i + 1);      // wraparound for n=14, i=13
        applyCNOT(i, t, n, sRe, sIm);
        __syncthreads();
    }
}

__device__ void ansatz(const float* w, int stageOff, int n, float* sRe, float* sIm, SMEM& sm) {
    for (int d = 0; d < REPSN; ++d)
        layer(w + d * WROW + stageOff, n, sRe, sIm, sm);
}

__device__ float blockSum(float v, SMEM& sm) {
    if (threadIdx.x == 0) sm.red = 0.f;
    __syncthreads();
    atomicAdd(&sm.red, v);        // ds_add_f32
    __syncthreads();
    float r = sm.red;
    __syncthreads();
    return r;
}

extern "C" __global__ __launch_bounds__(BLOCK)
void pqc_kernel(const float* __restrict__ sr, const float* __restrict__ si,
                const float* __restrict__ w, float* __restrict__ out) {
    extern __shared__ float dyn[];                 // 2 * 16384 f32 = 128 KB (fits 320 KB WGP LDS)
    float* sRe = dyn;
    float* sIm = dyn + D2;
    __shared__ SMEM sm;

    const int b = blockIdx.x;
    const float* srb = sr + (size_t)b * D1;
    const float* sib = si + (size_t)b * D1;
    for (int i = threadIdx.x; i < D1; i += BLOCK) { sRe[i] = srb[i]; sIm[i] = sib[i]; }
    __syncthreads();

    // Stage 1: 13-qubit ansatz (weights [:, :78])
    ansatz(w, 0, NQ, sRe, sIm, sm);

    // Ancilla extension: concat zeros -> 14-qubit state
    for (int i = threadIdx.x; i < D1; i += BLOCK) { sRe[D1 + i] = 0.f; sIm[D1 + i] = 0.f; }
    __syncthreads();

    // Stage 2: 14-qubit ansatz (weights [:, 78:156])
    ansatz(w, 78, NQ2, sRe, sIm, sm);

    // Truncate to first 8192, normalize, leaky-ReLU on re/im, renormalize
    float acc = 0.f;
    for (int i = threadIdx.x; i < D1; i += BLOCK) {
        float r = sRe[i], ii = sIm[i];
        acc += r * r + ii * ii;
    }
    float scale = 1.f / (sqrtf(blockSum(acc, sm)) + EPSV);
    float acc2 = 0.f;
    for (int i = threadIdx.x; i < D1; i += BLOCK) {
        float r  = sRe[i] * scale; r  = (r  >= 0.f) ? r  : NEG_SLOPE * r;
        float ii = sIm[i] * scale; ii = (ii >= 0.f) ? ii : NEG_SLOPE * ii;
        sRe[i] = r; sIm[i] = ii;
        acc2 += r * r + ii * ii;
    }
    float inv2 = 1.f / sqrtf(blockSum(acc2, sm));
    for (int i = threadIdx.x; i < D1; i += BLOCK) { sRe[i] *= inv2; sIm[i] *= inv2; }
    __syncthreads();

    // Stage 3: 13-qubit ansatz (weights [:, 156:234])
    ansatz(w, 156, NQ, sRe, sIm, sm);

    // Pauli-Z expectations per qubit
    float e[NQ];
#pragma unroll
    for (int k = 0; k < NQ; ++k) e[k] = 0.f;
    for (int i = threadIdx.x; i < D1; i += BLOCK) {
        float r = sRe[i], ii = sIm[i];
        float p = r * r + ii * ii;
#pragma unroll
        for (int k = 0; k < NQ; ++k)
            e[k] += ((i >> (NQ - 1 - k)) & 1) ? -p : p;
    }
    if (threadIdx.x < NQ) sm.ex[threadIdx.x] = 0.f;
    __syncthreads();
#pragma unroll
    for (int k = 0; k < NQ; ++k) atomicAdd(&sm.ex[k], e[k]);
    __syncthreads();
    if (threadIdx.x < NQ) out[(size_t)b * NQ + threadIdx.x] = sm.ex[threadIdx.x];
}

extern "C" void kernel_launch(void* const* d_in, const int* in_sizes, int n_in,
                              void* d_out, int out_size, void* d_ws, size_t ws_size,
                              hipStream_t stream) {
    (void)in_sizes; (void)n_in; (void)d_ws; (void)ws_size; (void)out_size;
    const float* sr = (const float*)d_in[0];
    const float* si = (const float*)d_in[1];
    const float* w  = (const float*)d_in[2];
    float* out = (float*)d_out;
    const size_t shmem = (size_t)2 * D2 * sizeof(float);   // 128 KB dynamic LDS
    (void)hipFuncSetAttribute(reinterpret_cast<const void*>(pqc_kernel),
                              hipFuncAttributeMaxDynamicSharedMemorySize, (int)shmem);
    pqc_kernel<<<256, BLOCK, shmem, stream>>>(sr, si, w, out);
}
